// ArcFaceLoss_82970178224588
// MI455X (gfx1250) — matrix operand order repaired and checked by reading, not compiled, not run on hardware
//
#include <hip/hip_runtime.h>
#include <math.h>

// ---------------- problem constants ----------------
#define B_ROWS   1024
#define D_FEAT   256
#define C_CLS    100000
#define NBLK     391                 // ceil(100000/256)
#define C_PAD    (NBLK * 256)        // 100096
#define S_SCALE  32.0f
#define COS_M_F  0.9553364891256061f
#define SIN_M_F  0.29552020666133955f
#define TH_F     (-0.9553364891256061f)
#define MM_F     0.08865606199840187f
#define EPS_F    1e-6f
#define CLAMP_F  1e-7f
#define NEG_BIG  (-3.0e38f)

typedef __attribute__((ext_vector_type(8)))  __bf16 v8bf;
typedef __attribute__((ext_vector_type(16))) __bf16 v16bf;
typedef __attribute__((ext_vector_type(8)))  float  v8f;

// ---------------- bf16 helpers (RNE) ----------------
static __device__ __forceinline__ unsigned short f2bf(float f) {
    unsigned int u = __float_as_uint(f);
    u += 0x7FFFu + ((u >> 16) & 1u);
    return (unsigned short)(u >> 16);
}
static __device__ __forceinline__ float bf2f(unsigned short h) {
    return __uint_as_float(((unsigned int)h) << 16);
}
static __device__ __forceinline__ v16bf ld_frag(const unsigned short* p0,
                                                const unsigned short* p1) {
    v8bf a = *reinterpret_cast<const v8bf*>(p0);
    v8bf b = *reinterpret_cast<const v8bf*>(p1);
    return __builtin_shufflevector(a, b, 0,1,2,3,4,5,6,7,8,9,10,11,12,13,14,15);
}
static __device__ __forceinline__ v8f wmma_bf16(v16bf a, v16bf b, v8f c) {
    return __builtin_amdgcn_wmma_f32_16x16x32_bf16(false, a, false, b,
                                                   (short)0, c, false, false);
}

// ---------------- K1/K2: row-normalize + hi/lo bf16 split ----------------
// grid.x = rows, block = 256 (one thread per feature)
__global__ __launch_bounds__(256)
void norm_split_kernel(const float* __restrict__ src,
                       unsigned short* __restrict__ dhi,
                       unsigned short* __restrict__ dlo) {
    const int row  = blockIdx.x;
    const int t    = threadIdx.x;
    const int wid  = t >> 5, lane = t & 31;
    const size_t base = (size_t)row * D_FEAT;
    float x  = src[base + t];
    float ss = x * x;
    #pragma unroll
    for (int m = 16; m; m >>= 1) ss += __shfl_xor(ss, m, 32);
    __shared__ float red[8];
    __shared__ float invn;
    if (lane == 0) red[wid] = ss;
    __syncthreads();
    if (t == 0) {
        float tot = 0.f;
        #pragma unroll
        for (int i = 0; i < 8; ++i) tot += red[i];
        invn = 1.0f / fmaxf(sqrtf(tot), EPS_F);
    }
    __syncthreads();
    float xn = x * invn;
    unsigned short h = f2bf(xn);
    unsigned short l = f2bf(xn - bf2f(h));
    dhi[base + t] = h;
    dlo[base + t] = l;
}

// ---------------- K3: exact fp32 cosine at the label class ----------------
// grid.x = B_ROWS, block = 256
__global__ __launch_bounds__(256)
void label_cos_kernel(const float* __restrict__ x,
                      const float* __restrict__ w,
                      const int*   __restrict__ labels,
                      float*       __restrict__ cosLab) {
    const int i = blockIdx.x;
    const int t = threadIdx.x;
    const int wid = t >> 5, lane = t & 31;
    const int l = labels[i];
    float xv = x[(size_t)i * D_FEAT + t];
    float wv = w[(size_t)l * D_FEAT + t];
    float sxx = xv * xv, sww = wv * wv, sxw = xv * wv;
    #pragma unroll
    for (int m = 16; m; m >>= 1) {
        sxx += __shfl_xor(sxx, m, 32);
        sww += __shfl_xor(sww, m, 32);
        sxw += __shfl_xor(sxw, m, 32);
    }
    __shared__ float rxx[8], rww[8], rxw[8];
    if (lane == 0) { rxx[wid] = sxx; rww[wid] = sww; rxw[wid] = sxw; }
    __syncthreads();
    if (t == 0) {
        float xx = 0.f, ww = 0.f, xw = 0.f;
        #pragma unroll
        for (int k = 0; k < 8; ++k) { xx += rxx[k]; ww += rww[k]; xw += rxw[k]; }
        float c = xw / (fmaxf(sqrtf(xx), EPS_F) * fmaxf(sqrtf(ww), EPS_F));
        c = fminf(fmaxf(c, -1.0f + CLAMP_F), 1.0f - CLAMP_F);
        cosLab[i] = c;
    }
}

// ---------------- K4: fused bf16x2 WMMA GEMM + per-tile logsumexp ----------------
// grid = (NBLK class-blocks of 256, 32 row-blocks of 32); block = 128 (4 waves).
// Each wave owns 32 rows x 64 classes: two 16-row M-tiles share every B fragment,
// so each (bh,bl) pair feeds 6 WMMAs -> ~65 FLOP/byte of L2 traffic.
// K-loop kept rolled (#pragma unroll 1) so live VGPRs stay < 256: no VGPR-MSB
// switching and ~6 waves/SIMD of occupancy to hide the gathered B-row L2 latency.
__global__ __launch_bounds__(128)
void cosine_lse_kernel(const unsigned short* __restrict__ xnHi,
                       const unsigned short* __restrict__ xnLo,
                       const unsigned short* __restrict__ wnHi,
                       const unsigned short* __restrict__ wnLo,
                       float* __restrict__ partM,
                       float* __restrict__ partS) {
    const int tid  = threadIdx.x;
    const int wid  = tid >> 5;
    const int lane = tid & 31;
    const int half = lane >> 4;
    const int idx  = lane & 15;
    const int cb      = blockIdx.x;
    const int rowBase = blockIdx.y * 32;
    const int colWave = cb * 256 + wid * 64;

    // A rows for the two M-tiles (M = idx for both lane halves)
    const unsigned short* aH0 = xnHi + (size_t)(rowBase + idx) * D_FEAT;
    const unsigned short* aL0 = xnLo + (size_t)(rowBase + idx) * D_FEAT;
    const unsigned short* aH1 = xnHi + (size_t)(rowBase + 16 + idx) * D_FEAT;
    const unsigned short* aL1 = xnLo + (size_t)(rowBase + 16 + idx) * D_FEAT;
    // B rows (one class per lane per N-tile); always < C_PAD so in-bounds
    const unsigned short* bH[4]; const unsigned short* bL[4];
    #pragma unroll
    for (int t = 0; t < 4; ++t) {
        const size_t c = (size_t)(colWave + t * 16 + idx);
        bH[t] = wnHi + c * D_FEAT;
        bL[t] = wnLo + c * D_FEAT;
    }

    v8f acc[2][4];
    #pragma unroll
    for (int rt = 0; rt < 2; ++rt)
        #pragma unroll
        for (int t = 0; t < 4; ++t)
            acc[rt][t] = (v8f){0.f,0.f,0.f,0.f,0.f,0.f,0.f,0.f};

    // K loop: 8 chunks of 32; per B frag pair: hi*hi + hi*lo + lo*hi for both M-tiles
    #pragma unroll 1
    for (int kc = 0; kc < 8; ++kc) {
        const int k0   = kc * 32;
        const int offA = k0 + half * 8;    // A: K interleaved by 8 per lane-half
        const int offB = k0 + half * 8;    // B: K interleaved by 8 per lane-half
        v16bf ah0 = ld_frag(aH0 + offA, aH0 + offA + 16);
        v16bf al0 = ld_frag(aL0 + offA, aL0 + offA + 16);
        v16bf ah1 = ld_frag(aH1 + offA, aH1 + offA + 16);
        v16bf al1 = ld_frag(aL1 + offA, aL1 + offA + 16);
        #pragma unroll
        for (int t = 0; t < 4; ++t) {
            v16bf bh = ld_frag(bH[t] + offB, bH[t] + offB + 16);
            v16bf bl = ld_frag(bL[t] + offB, bL[t] + offB + 16);
            acc[0][t] = wmma_bf16(ah0, bh, acc[0][t]);
            acc[0][t] = wmma_bf16(ah0, bl, acc[0][t]);
            acc[0][t] = wmma_bf16(al0, bh, acc[0][t]);
            acc[1][t] = wmma_bf16(ah1, bh, acc[1][t]);
            acc[1][t] = wmma_bf16(ah1, bl, acc[1][t]);
            acc[1][t] = wmma_bf16(al1, bh, acc[1][t]);
        }
    }

    // acc[rt][t][r] = cosine(row = rowBase + rt*16 + r + 8*half,
    //                        col = colWave + t*16 + idx)
    // Convert to masked logits in place, reduce per-row max and sum-exp.
    float mx[2][8], sm[2][8];
    #pragma unroll
    for (int rt = 0; rt < 2; ++rt) {
        #pragma unroll
        for (int r = 0; r < 8; ++r) mx[rt][r] = NEG_BIG;
        #pragma unroll
        for (int t = 0; t < 4; ++t) {
            const bool ok = (colWave + t * 16 + idx) < C_CLS;
            #pragma unroll
            for (int r = 0; r < 8; ++r) {
                float c = fminf(fmaxf(acc[rt][t][r], -1.0f + CLAMP_F), 1.0f - CLAMP_F);
                float lg = ok ? (S_SCALE * c) : NEG_BIG;
                acc[rt][t][r] = lg;
                mx[rt][r] = fmaxf(mx[rt][r], lg);
            }
        }
        #pragma unroll
        for (int r = 0; r < 8; ++r) {
            #pragma unroll
            for (int m = 8; m; m >>= 1)
                mx[rt][r] = fmaxf(mx[rt][r], __shfl_xor(mx[rt][r], m, 32));
        }
        #pragma unroll
        for (int r = 0; r < 8; ++r) {
            float p = 0.f;
            #pragma unroll
            for (int t = 0; t < 4; ++t) p += __expf(acc[rt][t][r] - mx[rt][r]);
            #pragma unroll
            for (int m = 8; m; m >>= 1) p += __shfl_xor(p, m, 32);
            sm[rt][r] = p;
        }
    }

    // Combine the 4 waves' (m,s) per row via LDS, emit per-(row, class-block) partials.
    __shared__ float shM[4][32], shS[4][32];
    if (idx == 0) {
        #pragma unroll
        for (int rt = 0; rt < 2; ++rt)
            #pragma unroll
            for (int r = 0; r < 8; ++r) {
                shM[wid][rt * 16 + half * 8 + r] = mx[rt][r];
                shS[wid][rt * 16 + half * 8 + r] = sm[rt][r];
            }
    }
    __syncthreads();
    if (tid < 32) {
        float M = NEG_BIG, S = 0.f;
        #pragma unroll
        for (int w = 0; w < 4; ++w) {
            float mw = shM[w][tid], sw = shS[w][tid];
            if (mw > M) { S = S * __expf(M - mw) + sw; M = mw; }
            else        { S += sw * __expf(mw - M); }
        }
        const size_t o = (size_t)cb * B_ROWS + rowBase + tid;
        partM[o] = M;
        partS[o] = S;
    }
}

// ---------------- K5: reduce partials, label fix-up, mean NLL ----------------
// single block of 256 threads
__global__ __launch_bounds__(256)
void finalize_kernel(const float* __restrict__ partM,
                     const float* __restrict__ partS,
                     const float* __restrict__ cosLab,
                     float* __restrict__ out) {
    const int t = threadIdx.x;
    float acc = 0.f;
    for (int i = t; i < B_ROWS; i += 256) {
        float m = NEG_BIG, s = 0.f;
        for (int b = 0; b < NBLK; ++b) {
            float mb = partM[(size_t)b * B_ROWS + i];
            float sb = partS[(size_t)b * B_ROWS + i];
            if (mb > m) { s = s * __expf(m - mb) + sb; m = mb; }
            else        { s += sb * __expf(mb - m); }
        }
        float lse = m + logf(s);
        float c   = cosLab[i];
        float sine = sqrtf(fmaxf(1.0f - c * c, 0.0f));
        float phi  = c * COS_M_F - sine * SIN_M_F;
        phi = (c > TH_F) ? phi : (c - MM_F);
        float lphi = S_SCALE * phi, lcos = S_SCALE * c;
        // replace the label-class term exp(S*cos) by exp(S*phi) inside the sum
        float d = expf(lphi - lse) - expf(lcos - lse);
        float lseN = lse + log1pf(d);
        acc += lseN - lphi;
    }
    __shared__ float red[256];
    red[t] = acc;
    __syncthreads();
    for (int sft = 128; sft; sft >>= 1) {
        if (t < sft) red[t] += red[t + sft];
        __syncthreads();
    }
    if (t == 0) out[0] = red[0] * (1.0f / (float)B_ROWS);
}

// ---------------- host-side launcher ----------------
extern "C" void kernel_launch(void* const* d_in, const int* in_sizes, int n_in,
                              void* d_out, int out_size, void* d_ws, size_t ws_size,
                              hipStream_t stream) {
    const float* emb    = (const float*)d_in[0];   // [1024, 256]
    const float* weight = (const float*)d_in[1];   // [100000, 256]
    const int*   labels = (const int*)d_in[2];     // [1024]
    float* out = (float*)d_out;

    // workspace layout (all 16B aligned)
    char* ws = (char*)d_ws;
    size_t off = 0;
    unsigned short* wnHi = (unsigned short*)(ws + off); off += (size_t)C_PAD * D_FEAT * 2;
    unsigned short* wnLo = (unsigned short*)(ws + off); off += (size_t)C_PAD * D_FEAT * 2;
    unsigned short* xnHi = (unsigned short*)(ws + off); off += (size_t)B_ROWS * D_FEAT * 2;
    unsigned short* xnLo = (unsigned short*)(ws + off); off += (size_t)B_ROWS * D_FEAT * 2;
    float* cosLab = (float*)(ws + off); off += (size_t)B_ROWS * 4;
    float* partM  = (float*)(ws + off); off += (size_t)NBLK * B_ROWS * 4;
    float* partS  = (float*)(ws + off); off += (size_t)NBLK * B_ROWS * 4;
    (void)ws_size; (void)in_sizes; (void)n_in; (void)out_size;

    hipLaunchKernelGGL(norm_split_kernel, dim3(B_ROWS), dim3(256), 0, stream,
                       emb, xnHi, xnLo);
    hipLaunchKernelGGL(norm_split_kernel, dim3(C_CLS), dim3(256), 0, stream,
                       weight, wnHi, wnLo);
    hipLaunchKernelGGL(label_cos_kernel, dim3(B_ROWS), dim3(256), 0, stream,
                       emb, weight, labels, cosLab);
    hipLaunchKernelGGL(cosine_lse_kernel, dim3(NBLK, B_ROWS / 32), dim3(128), 0, stream,
                       xnHi, xnLo, wnHi, wnLo, partM, partS);
    hipLaunchKernelGGL(finalize_kernel, dim3(1), dim3(256), 0, stream,
                       partM, partS, cosLab, out);
}
